// Self_Attention_66013647340270
// MI455X (gfx1250) — compile-verified, matrix operand order checked by reading, not run on hardware
//
#include <hip/hip_runtime.h>

// Shapes fixed by the reference
#define BATCH 4
#define CCH   256     // channels
#define NTOK  4096    // H*W
#define EMB   32      // C/8

typedef __attribute__((ext_vector_type(16))) _Float16 v16h;
typedef __attribute__((ext_vector_type(8)))  _Float16 v8h;
typedef __attribute__((ext_vector_type(8)))  float    v8f;

// swap value with lane^16 (wave32): ds_swizzle group-of-32, xor=0x10, and=0x1f
static __device__ __forceinline__ float xor16(float v) {
    return __int_as_float(__builtin_amdgcn_ds_swizzle(__float_as_int(v), 0x401F));
}

// Load a 16x16x32-f16 WMMA A/B operand fragment for this lane.
// p points at a 32-halfword row (fT/gT row, or hh row + i0). off0 = (lane<16 ? 0 : 8).
// Per the CDNA5 16-bit A layout: lanes 0-15 hold K 0-7 & 16-23, lanes 16-31 hold K 8-15 & 24-31.
static __device__ __forceinline__ v16h load_frag(const _Float16* p, int off0) {
    v8h lo = *(const v8h*)(p + off0);
    v8h hi = *(const v8h*)(p + off0 + 16);
    return __builtin_shufflevector(lo, hi, 0,1,2,3,4,5,6,7,8,9,10,11,12,13,14,15);
}

// ---------------------------------------------------------------------------
// Kernel 1: fused key/query/value 1x1 conv projections.
//   fT [B][N][E] f16  (A operand of S = f^T g, row = token)
//   gT [B][N][E] f16  (B operand of S, row = token)
//   hE [B][E][N] f16  (A operand of P*V, row = channel)
// ---------------------------------------------------------------------------
__global__ __launch_bounds__(256) void proj_kernel(
    const float* __restrict__ x,
    const float* __restrict__ kw, const float* __restrict__ kb,
    const float* __restrict__ qw, const float* __restrict__ qb,
    const float* __restrict__ vw, const float* __restrict__ vb,
    _Float16* __restrict__ fT, _Float16* __restrict__ gT, _Float16* __restrict__ hE)
{
    const int bb = blockIdx.x >> 4;
    const int n  = ((blockIdx.x & 15) << 8) + threadIdx.x;

    float af[EMB], ag[EMB], ah[EMB];
    #pragma unroll
    for (int e = 0; e < EMB; ++e) { af[e] = kb[e]; ag[e] = qb[e]; ah[e] = vb[e]; }

    const float* xp = x + (size_t)bb * CCH * NTOK + n;
    for (int c = 0; c < CCH; ++c) {
        float xv = xp[(size_t)c * NTOK];          // coalesced across threads
        #pragma unroll
        for (int e = 0; e < EMB; ++e) {           // weights are wave-uniform (scalar path)
            af[e] = fmaf(kw[e * CCH + c], xv, af[e]);
            ag[e] = fmaf(qw[e * CCH + c], xv, ag[e]);
            ah[e] = fmaf(vw[e * CCH + c], xv, ah[e]);
        }
    }

    _Float16* fp = fT + ((size_t)bb * NTOK + n) * EMB;
    _Float16* gp = gT + ((size_t)bb * NTOK + n) * EMB;
    _Float16* hp = hE + (size_t)bb * EMB * NTOK + n;
    #pragma unroll
    for (int e = 0; e < EMB; ++e) {
        fp[e] = (_Float16)af[e];
        gp[e] = (_Float16)ag[e];
        hp[(size_t)e * NTOK] = (_Float16)ah[e];
    }
}

// ---------------------------------------------------------------------------
// Kernel 2: flash attention over the key axis (softmax axis=1 == i axis).
// One wave handles TWO 16-column j tiles (32 columns): the fT / hE A-operand
// fragments are shared by both tiles -> 8 WMMAs per i-block, half the L2
// traffic of the 1-tile version. Online softmax, all GEMMs via v_wmma.
// ---------------------------------------------------------------------------
__global__ __launch_bounds__(128) void attn_kernel(
    const _Float16* __restrict__ fT, const _Float16* __restrict__ gT,
    const _Float16* __restrict__ hE, float* __restrict__ vO)
{
    const int lane  = threadIdx.x & 31;
    const int bb    = blockIdx.x >> 5;                               // batch
    const int jpair = ((blockIdx.x & 31) << 2) + (threadIdx.x >> 5); // 32-col tile
    const int j0    = jpair << 5;
    const int half  = lane >> 4;
    const int m16   = lane & 15;
    const int off0  = half << 3;   // 0 or 8 (K-chunk select per A/B layout)

    // B operands (queries g for our 2x16 columns) are loop invariant.
    v8f acc0[2], acc1[2];          // v rows e=0..15 / e=16..31 per j tile
    v16h bg[2];
    float mrun[2], lrun[2];
    #pragma unroll
    for (int t = 0; t < 2; ++t) {
        bg[t]   = load_frag(gT + ((size_t)bb * NTOK + (j0 + 16 * t + m16)) * EMB, off0);
        acc0[t] = (v8f){};
        acc1[t] = (v8f){};
        mrun[t] = -__builtin_inff();
        lrun[t] = 0.f;             // per-lane partial; lane-pair combined at end
    }

    const _Float16* fBase = fT + (size_t)bb * NTOK * EMB;
    const _Float16* hBase = hE + (size_t)bb * EMB * NTOK;

    for (int ib = 0; ib < NTOK / 32; ++ib) {
        const int i0 = ib << 5;
        // WGP-scope prefetch of the next i-block (pull into all cache levels)
        __builtin_prefetch(fBase + (size_t)(i0 + 32 + m16) * EMB, 0, 3);
        __builtin_prefetch(hBase + (size_t)m16 * NTOK + i0 + 32, 0, 3);

        // Shared A fragments: f^T rows i0..i0+15 and i0+16..i0+31
        v16h af0 = load_frag(fBase + (size_t)(i0 + m16)      * EMB, off0);
        v16h af1 = load_frag(fBase + (size_t)(i0 + 16 + m16) * EMB, off0);

        // S tiles: s[i-block][j tile t]
        v8f z = {};
        v8f s0[2], s1[2];
        s0[0] = __builtin_amdgcn_wmma_f32_16x16x32_f16(false, af0, false, bg[0], (short)0, z, false, false);
        s0[1] = __builtin_amdgcn_wmma_f32_16x16x32_f16(false, af0, false, bg[1], (short)0, z, false, false);
        s1[0] = __builtin_amdgcn_wmma_f32_16x16x32_f16(false, af1, false, bg[0], (short)0, z, false, false);
        s1[1] = __builtin_amdgcn_wmma_f32_16x16x32_f16(false, af1, false, bg[1], (short)0, z, false, false);

        // Shared A fragments for PV: hh rows e and e+16, K = this i-block
        v16h ah0 = load_frag(hBase + (size_t)m16        * NTOK + i0, off0);
        v16h ah1 = load_frag(hBase + (size_t)(m16 + 16) * NTOK + i0, off0);

        v16h bp[2];
        #pragma unroll
        for (int t = 0; t < 2; ++t) {
            // Per-column (per j) max: 16 in-lane values + lane-pair swap.
            float mloc = s0[t][0];
            #pragma unroll
            for (int r = 1; r < 8; ++r) mloc = fmaxf(mloc, s0[t][r]);
            #pragma unroll
            for (int r = 0; r < 8; ++r) mloc = fmaxf(mloc, s1[t][r]);
            mloc = fmaxf(mloc, xor16(mloc));

            const float mnew  = fmaxf(mrun[t], mloc);
            const float alpha = __expf(mrun[t] - mnew);   // 0 on first iteration
            mrun[t] = mnew;
            lrun[t] *= alpha;
            acc0[t] *= alpha;
            acc1[t] *= alpha;

            v8f p0, p1;
            float ls = 0.f;
            #pragma unroll
            for (int r = 0; r < 8; ++r) { p0[r] = __expf(s0[t][r] - mnew); ls += p0[r]; }
            #pragma unroll
            for (int r = 0; r < 8; ++r) { p1[r] = __expf(s1[t][r] - mnew); ls += p1[r]; }
            lrun[t] += ls;

            // D-layout of (P0,P1) converts lane-locally to the B operand layout:
            // lanes 0-15 hold K 0-7 (P0 rows) & 16-23 (P1 rows); lanes 16-31: 8-15 & 24-31.
            #pragma unroll
            for (int r = 0; r < 8; ++r) {
                bp[t][r]     = (_Float16)p0[r];
                bp[t][8 + r] = (_Float16)p1[r];
            }
        }

        // PV: acc[e, j] += hh[e, i-block] * P[i-block, j]
        acc0[0] = __builtin_amdgcn_wmma_f32_16x16x32_f16(false, ah0, false, bp[0], (short)0, acc0[0], false, false);
        acc0[1] = __builtin_amdgcn_wmma_f32_16x16x32_f16(false, ah0, false, bp[1], (short)0, acc0[1], false, false);
        acc1[0] = __builtin_amdgcn_wmma_f32_16x16x32_f16(false, ah1, false, bp[0], (short)0, acc1[0], false, false);
        acc1[1] = __builtin_amdgcn_wmma_f32_16x16x32_f16(false, ah1, false, bp[1], (short)0, acc1[1], false, false);
    }

    const int e0 = half << 3;   // D layout: lanes 0-15 rows 0-7, lanes 16-31 rows 8-15
    #pragma unroll
    for (int t = 0; t < 2; ++t) {
        const float ltot = lrun[t] + xor16(lrun[t]);
        const float inv  = 1.f / ltot;
        const int jcol   = j0 + 16 * t + m16;
        float* vp = vO + (size_t)bb * EMB * NTOK + jcol;
        #pragma unroll
        for (int r = 0; r < 8; ++r) {
            vp[(size_t)(e0 + r)      * NTOK] = acc0[t][r] * inv;
            vp[(size_t)(e0 + r + 16) * NTOK] = acc1[t][r] * inv;
        }
    }
}

// ---------------------------------------------------------------------------
// Kernel 3: o = att_w @ v + att_b ; y = gamma*o + x. Writes both tuple outputs.
// ---------------------------------------------------------------------------
__global__ __launch_bounds__(256) void out_kernel(
    const float* __restrict__ vO, const float* __restrict__ aw,
    const float* __restrict__ ab, const float* __restrict__ x,
    const float* __restrict__ gammap,
    float* __restrict__ y, float* __restrict__ o)
{
    const int bb = blockIdx.x >> 4;
    const int n  = ((blockIdx.x & 15) << 8) + threadIdx.x;
    const float gamma = gammap[0];

    float vv[EMB];
    const float* vp = vO + (size_t)bb * EMB * NTOK + n;
    #pragma unroll
    for (int e = 0; e < EMB; ++e) vv[e] = vp[(size_t)e * NTOK];

    for (int c = 0; c < CCH; ++c) {
        float acc = ab[c];
        #pragma unroll
        for (int e = 0; e < EMB; ++e) acc = fmaf(aw[c * EMB + e], vv[e], acc);
        const size_t idx = ((size_t)bb * CCH + c) * NTOK + n;
        o[idx] = acc;
        y[idx] = fmaf(gamma, acc, x[idx]);
    }
}

extern "C" void kernel_launch(void* const* d_in, const int* in_sizes, int n_in,
                              void* d_out, int out_size, void* d_ws, size_t ws_size,
                              hipStream_t stream) {
    const float* x  = (const float*)d_in[0];
    const float* kw = (const float*)d_in[1];
    const float* kb = (const float*)d_in[2];
    const float* qw = (const float*)d_in[3];
    const float* qb = (const float*)d_in[4];
    const float* vw = (const float*)d_in[5];
    const float* vb = (const float*)d_in[6];
    const float* aw = (const float*)d_in[7];
    const float* ab = (const float*)d_in[8];
    const float* gm = (const float*)d_in[9];

    // Workspace: fT(1MB) | gT(1MB) | hE(1MB) | vO(2MB) = 5MB total
    char* ws = (char*)d_ws;
    _Float16* fT = (_Float16*)(ws);
    _Float16* gT = (_Float16*)(ws + (1u << 20));
    _Float16* hE = (_Float16*)(ws + (2u << 20));
    float*    vO = (float*)   (ws + (3u << 20));

    float* y = (float*)d_out;                         // tuple output 0
    float* o = y + (size_t)BATCH * CCH * NTOK;        // tuple output 1

    proj_kernel<<<BATCH * (NTOK / 256), 256, 0, stream>>>(x, kw, kb, qw, qb, vw, vb, fT, gT, hE);
    // 32 columns per wave, 4 waves per block
    attn_kernel<<<BATCH * (NTOK / 32 / 4), 128, 0, stream>>>(fT, gT, hE, vO);
    out_kernel <<<BATCH * (NTOK / 256), 256, 0, stream>>>(vO, aw, ab, x, gm, y, o);
}
